// EmbeddedAEForcastingQuant_54623394070855
// MI455X (gfx1250) — matrix operand order repaired and checked by reading, not compiled, not run on hardware
//
#include <hip/hip_runtime.h>
#include <hip/hip_bf16.h>
#include <math.h>

// ---------------------------------------------------------------------------
// Problem constants (from reference)
// ---------------------------------------------------------------------------
#define B_    2048
#define F_    32
#define H_    512
#define BN_   64
#define OUT_  8
#define SEQ_  128
#define LA_   32
#define TWOH  (2 * H_)
#define FOURH (4 * H_)
#define XLD   ((SEQ_ + LA_) * F_)   // 5120, row stride of x
#define OLD   ((1 + LA_) * OUT_)    // 264, row stride of d_out

typedef __attribute__((ext_vector_type(16))) __bf16 v16bf;
typedef __attribute__((ext_vector_type(8)))  __bf16 v8bf;
typedef __attribute__((ext_vector_type(8)))  float  v8f;

#define LSTR 48   // LDS row stride in bf16 elems (96B: 16B-aligned, conflict-friendly)

__device__ __forceinline__ unsigned lds_off32(const void* p) {
    // gfx1250 flat->LDS aperture mapping: LDS offset is the low 32 bits of the address
    return (unsigned)(size_t)p;
}

// ---------------------------------------------------------------------------
// Generic bf16 WMMA GEMM:  C[M,N] = A1[M,K1] @ B1t^T + A2[M,K2] @ B2t^T + bias
// B matrices pre-transposed: Bt[n][k] (row stride = K of that pass).
// Tile 128x128 / block (8 waves; wave = 64x32 = 4x2 WMMA tiles), BK=32.
// Double-buffered LDS filled by async direct-to-LDS copies (ASYNCcnt),
// prefetching K-step i+1 while WMMAs consume step i.
// mode 0: v = acc+bias -> Cf (f32) and/or Cb (bf16 mirror)
// mode 1: v = rint(tanh(acc+bias)) -> Cb (AE quantizer)
// M multiple of 128 (2048 here). N guarded on store; B rows clamped on load
// (clamped rows only feed columns that the guard discards).
// ---------------------------------------------------------------------------
__global__ void __launch_bounds__(256)
gemm_wmma_bf16(const __bf16* __restrict__ A1, int lda1,
               const __bf16* __restrict__ B1t, int K1,
               const __bf16* __restrict__ A2, int lda2,
               const __bf16* __restrict__ B2t, int K2,
               const float* __restrict__ bias, int mode,
               float* __restrict__ Cf, int ldc,
               __bf16* __restrict__ Cb, int ldcb, int N)
{
    __shared__ __bf16 Abuf[2][128 * LSTR];
    __shared__ __bf16 Bbuf[2][128 * LSTR];

    const int tid  = threadIdx.x;
    const int lane = tid & 31;
    const int wave = tid >> 5;          // 0..7
    const int wm   = wave >> 2;         // 0..1  (M dir, 64 rows each)
    const int wn   = wave & 3;          // 0..3  (N dir, 32 cols each)
    const int rr   = lane & 15;
    const int hf   = lane >> 4;

    const int mBase = blockIdx.y * 128;
    const int nBase = blockIdx.x * 128;

    // per-thread staging slot: 2 threads/row, 32 bytes each (two b128 chunks)
    const int sr   = tid >> 1;              // staged row 0..127
    const int scol = (tid & 1) << 4;        // 0 or 16 (bf16 elems)
    int nClamp = nBase + sr;                // B row (clamped below)
    if (nClamp > N - 1) nClamp = N - 1;

    const int it1 = K1 >> 5;
    const int it2 = K2 >> 5;
    const int itTot = it1 + it2;

    v8f acc[4][2];
#pragma unroll
    for (int tm = 0; tm < 4; ++tm)
#pragma unroll
        for (int tn = 0; tn < 2; ++tn)
            acc[tm][tn] = (v8f)(0.0f);

    // Issue the 4 async direct-to-LDS B128 copies for K-step `it` into buffer `buf`.
    // INST_OFFSET is added to BOTH the LDS and global address (ISA 08, §4.4),
    // so one base covers both 16B chunks of the staged 32B row-piece.
    auto issue = [&](int it, int buf) {
        const __bf16* A;
        const __bf16* Bt;
        int lda, Kp, k0;
        if (it < it1) { A = A1; Bt = B1t; lda = lda1; Kp = K1; k0 = it << 5; }
        else          { A = A2; Bt = B2t; lda = lda2; Kp = K2; k0 = (it - it1) << 5; }

        const __bf16* ag = A + (size_t)(mBase + sr) * lda + k0 + scol;
        unsigned      al = lds_off32(&Abuf[buf][sr * LSTR + scol]);
        asm volatile("global_load_async_to_lds_b128 %0, %1, off\n\t"
                     "global_load_async_to_lds_b128 %0, %1, off offset:16"
                     :: "v"(al), "v"(ag) : "memory");

        const __bf16* bg = Bt + (size_t)nClamp * Kp + k0 + scol;
        unsigned      bl = lds_off32(&Bbuf[buf][sr * LSTR + scol]);
        asm volatile("global_load_async_to_lds_b128 %0, %1, off\n\t"
                     "global_load_async_to_lds_b128 %0, %1, off offset:16"
                     :: "v"(bl), "v"(bg) : "memory");
    };

    issue(0, 0);   // prologue prefetch

    for (int it = 0; it < itTot; ++it) {
        const int cur = it & 1;

        // current buffer's async copies (ours) done; barrier makes everyone's
        // visible AND orders last iteration's LDS reads before buffer reuse.
        asm volatile("s_wait_asynccnt 0x0" ::: "memory");
        __syncthreads();

        if (it + 1 < itTot) issue(it + 1, cur ^ 1);   // overlaps with WMMAs below

        // ---- fragments (documented 16-bit A 16x32 / B 32x16 layouts) ----
        union Frag { v16bf v; v8bf h[2]; };
        Frag af[4], bfm[2];
#pragma unroll
        for (int tm = 0; tm < 4; ++tm) {
            // lane rr = row M; elems 0-7: K = hf*8..+7 ; elems 8-15: K = 16+hf*8..+7
            const __bf16* p = &Abuf[cur][(wm * 64 + tm * 16 + rr) * LSTR + hf * 8];
            af[tm].h[0] = *(const v8bf*)p;
            af[tm].h[1] = *(const v8bf*)(p + 16);
        }
#pragma unroll
        for (int tn = 0; tn < 2; ++tn) {
            // lane rr = col N; elems 0-7: K = hf*8..+7 ; elems 8-15: K = 16+hf*8..+7
            const __bf16* p = &Bbuf[cur][(wn * 32 + tn * 16 + rr) * LSTR + hf * 8];
            bfm[tn].h[0] = *(const v8bf*)p;
            bfm[tn].h[1] = *(const v8bf*)(p + 16);
        }

#pragma unroll
        for (int tm = 0; tm < 4; ++tm)
#pragma unroll
            for (int tn = 0; tn < 2; ++tn)
                acc[tm][tn] = __builtin_amdgcn_wmma_f32_16x16x32_bf16(
                    false, af[tm].v, false, bfm[tn].v,
                    (short)0, acc[tm][tn], false, false);
    }

    // ---- epilogue: C layout: VGPR j -> row = base + hf*8 + j, col = base + rr
#pragma unroll
    for (int tm = 0; tm < 4; ++tm)
#pragma unroll
        for (int tn = 0; tn < 2; ++tn) {
            const int col = nBase + wn * 32 + tn * 16 + rr;
            if (col >= N) continue;
            const int row0 = mBase + wm * 64 + tm * 16 + hf * 8;
            const float bv = bias ? bias[col] : 0.0f;
#pragma unroll
            for (int j = 0; j < 8; ++j) {
                float v = acc[tm][tn][j] + bv;
                if (mode == 1) v = rintf(tanhf(v));
                const int row = row0 + j;
                if (Cf) Cf[(size_t)row * ldc + col] = v;
                if (Cb) Cb[(size_t)row * ldcb + col] = (__bf16)v;
            }
        }
}

// ---------------------------------------------------------------------------
// Setup / pointwise kernels
// ---------------------------------------------------------------------------
__global__ void wt_cvt(const float* __restrict__ W, __bf16* __restrict__ Wt,
                       int K, int N)   // W[K][N] f32 -> Wt[N][K] bf16
{
    int i = blockIdx.x * 256 + threadIdx.x;
    if (i < K * N) {
        int n = i / K, k = i - n * K;
        Wt[i] = (__bf16)W[(size_t)k * N + n];
    }
}

__global__ void bias_combine(const float* __restrict__ a,
                             const float* __restrict__ b,
                             float* __restrict__ o, int n)
{
    int i = blockIdx.x * 256 + threadIdx.x;
    if (i < n) o[i] = a[i] + b[i];
}

__global__ void zero_state(float* __restrict__ hc, __bf16* __restrict__ hcb, int n)
{
    int i = blockIdx.x * 256 + threadIdx.x;
    if (i < n) { hc[i] = 0.0f; hcb[i] = (__bf16)0.0f; }
}

__device__ __forceinline__ float sigf(float x) { return 1.0f / (1.0f + __expf(-x)); }

// gates g[B,4H] (i|f|g|o), cprev from decoded AE; writes hc=[h|c] f32 + bf16 mirror
__global__ void lstm_pointwise(const float* __restrict__ g,
                               const float* __restrict__ cprev, int ldcp,
                               float* __restrict__ hc, __bf16* __restrict__ hcb)
{
    int idx = blockIdx.x * 256 + threadIdx.x;     // B*H threads
    if (idx >= B_ * H_) return;
    int m = idx / H_, j = idx - m * H_;
    const float* gr = g + (size_t)m * FOURH;
    float ig = sigf(gr[j]);
    float fg = sigf(gr[H_ + j]);
    float gg = tanhf(gr[2 * H_ + j]);
    float og = sigf(gr[3 * H_ + j]);
    float c2 = fg * cprev[(size_t)m * ldcp + j] + ig * gg;
    float h  = og * tanhf(c2);
    size_t o = (size_t)m * TWOH;
    hc[o + j]        = h;   hc[o + H_ + j]  = c2;
    hcb[o + j]       = (__bf16)h;
    hcb[o + H_ + j]  = (__bf16)c2;
}

// Build x_t (bf16). In look-ahead, columns [0,8) come from the previous output.
__global__ void xt_build(const float* __restrict__ x, int xoff,
                         const float* __restrict__ outPrev,
                         __bf16* __restrict__ xt)
{
    int idx = blockIdx.x * 256 + threadIdx.x;     // B*F threads
    if (idx >= B_ * F_) return;
    int m = idx >> 5, c = idx & 31;
    float v;
    if (outPrev && c < OUT_) v = outPrev[(size_t)m * OLD + c];
    else                     v = x[(size_t)m * XLD + xoff + c];
    xt[idx] = (__bf16)v;
}

// out[m,n] = h1[m,:] @ Wf[:,n] + bf ; h1 = cols [0,H) of hc1 (ld = 2H)
__global__ void dense_head(const float* __restrict__ hc1,
                           const float* __restrict__ Wf,
                           const float* __restrict__ bf_,
                           float* __restrict__ outp)
{
    int idx = blockIdx.x * 256 + threadIdx.x;     // B*OUT threads
    if (idx >= B_ * OUT_) return;
    int m = idx >> 3, n = idx & 7;
    const float* hr = hc1 + (size_t)m * TWOH;
    float s = bf_[n];
    for (int k = 0; k < H_; ++k) s = fmaf(hr[k], Wf[k * OUT_ + n], s);
    outp[(size_t)m * OLD + n] = s;
}

// ---------------------------------------------------------------------------
// Host orchestration
// ---------------------------------------------------------------------------
static inline void* ws_take(char*& w, size_t bytes) {
    void* p = (void*)w;
    w += (bytes + 255) & ~(size_t)255;
    return p;
}

extern "C" void kernel_launch(void* const* d_in, const int* in_sizes, int n_in,
                              void* d_out, int out_size, void* d_ws, size_t ws_size,
                              hipStream_t stream)
{
    (void)in_sizes; (void)n_in; (void)out_size; (void)ws_size;

    const float* x    = (const float*)d_in[0];
    const float* Wih0 = (const float*)d_in[1];
    const float* Whh0 = (const float*)d_in[2];
    const float* bih0 = (const float*)d_in[3];
    const float* bhh0 = (const float*)d_in[4];
    const float* Wih1 = (const float*)d_in[5];
    const float* Whh1 = (const float*)d_in[6];
    const float* bih1 = (const float*)d_in[7];
    const float* bhh1 = (const float*)d_in[8];
    const float* We   = (const float*)d_in[9];
    const float* be   = (const float*)d_in[10];
    const float* Wd   = (const float*)d_in[11];
    const float* bd   = (const float*)d_in[12];
    const float* Wf   = (const float*)d_in[13];
    const float* bfv  = (const float*)d_in[14];
    float* outp = (float*)d_out;

    char* w = (char*)d_ws;
    __bf16* Wih0T = (__bf16*)ws_take(w, (size_t)FOURH * F_  * 2);
    __bf16* Whh0T = (__bf16*)ws_take(w, (size_t)FOURH * H_  * 2);
    __bf16* Wih1T = (__bf16*)ws_take(w, (size_t)FOURH * H_  * 2);
    __bf16* Whh1T = (__bf16*)ws_take(w, (size_t)FOURH * H_  * 2);
    __bf16* WeT   = (__bf16*)ws_take(w, (size_t)BN_  * TWOH * 2);
    __bf16* WdT   = (__bf16*)ws_take(w, (size_t)TWOH * BN_  * 2);
    float*  b0    = (float*) ws_take(w, (size_t)FOURH * 4);
    float*  b1    = (float*) ws_take(w, (size_t)FOURH * 4);
    float*  hc0   = (float*) ws_take(w, (size_t)B_ * TWOH * 4);
    __bf16* hc0b  = (__bf16*)ws_take(w, (size_t)B_ * TWOH * 2);
    float*  hc1   = (float*) ws_take(w, (size_t)B_ * TWOH * 4);
    __bf16* hc1b  = (__bf16*)ws_take(w, (size_t)B_ * TWOH * 2);
    float*  hcd   = (float*) ws_take(w, (size_t)B_ * TWOH * 4);
    __bf16* hcdb  = (__bf16*)ws_take(w, (size_t)B_ * TWOH * 2);
    __bf16* q     = (__bf16*)ws_take(w, (size_t)B_ * BN_  * 2);
    float*  g     = (float*) ws_take(w, (size_t)B_ * FOURH * 4);
    __bf16* xt    = (__bf16*)ws_take(w, (size_t)B_ * F_   * 2);

    const int TB = 256;
    #define CDIV(a, b) (((a) + (b) - 1) / (b))

    // ---- one-time setup (inside graph; deterministic & cheap) ----
    wt_cvt<<<CDIV(F_  * FOURH, TB), TB, 0, stream>>>(Wih0, Wih0T, F_,  FOURH);
    wt_cvt<<<CDIV(H_  * FOURH, TB), TB, 0, stream>>>(Whh0, Whh0T, H_,  FOURH);
    wt_cvt<<<CDIV(H_  * FOURH, TB), TB, 0, stream>>>(Wih1, Wih1T, H_,  FOURH);
    wt_cvt<<<CDIV(H_  * FOURH, TB), TB, 0, stream>>>(Whh1, Whh1T, H_,  FOURH);
    wt_cvt<<<CDIV(TWOH * BN_,  TB), TB, 0, stream>>>(We,   WeT,   TWOH, BN_);
    wt_cvt<<<CDIV(BN_ * TWOH,  TB), TB, 0, stream>>>(Wd,   WdT,   BN_,  TWOH);
    bias_combine<<<CDIV(FOURH, TB), TB, 0, stream>>>(bih0, bhh0, b0, FOURH);
    bias_combine<<<CDIV(FOURH, TB), TB, 0, stream>>>(bih1, bhh1, b1, FOURH);
    zero_state<<<CDIV(B_ * TWOH, TB), TB, 0, stream>>>(hc0, hc0b, B_ * TWOH);
    zero_state<<<CDIV(B_ * TWOH, TB), TB, 0, stream>>>(hc1, hc1b, B_ * TWOH);

    const dim3 gGates(FOURH / 128, B_ / 128);      // 16 x 16
    const dim3 gDec(TWOH / 128,   B_ / 128);       //  8 x 16
    const dim3 gEnc(1,            B_ / 128);       //  1 x 16

    for (int t = 0; t < SEQ_ + LA_; ++t) {
        const bool la = (t >= SEQ_);

        // x_t (bf16); in look-ahead, splice previous output into cols [0,8)
        xt_build<<<CDIV(B_ * F_, TB), TB, 0, stream>>>(
            x, t * F_, la ? (outp + (size_t)(t - SEQ_) * OUT_) : nullptr, xt);

        // -------- layer 0 --------
        // AE encode+quantize: q = round(tanh(hc0 @ We + be))
        gemm_wmma_bf16<<<gEnc, TB, 0, stream>>>(
            hc0b, TWOH, WeT, TWOH, nullptr, 0, nullptr, 0,
            be, /*mode=*/1, nullptr, 0, q, BN_, BN_);
        // AE decode: hcd = q @ Wd + bd  (f32 + bf16 mirror)
        gemm_wmma_bf16<<<gDec, TB, 0, stream>>>(
            q, BN_, WdT, BN_, nullptr, 0, nullptr, 0,
            bd, /*mode=*/0, hcd, TWOH, hcdb, TWOH, TWOH);
        // gates: g = x_t @ Wih0 + h0' @ Whh0 + b0
        gemm_wmma_bf16<<<gGates, TB, 0, stream>>>(
            xt, F_, Wih0T, F_, hcdb, TWOH, Whh0T, H_,
            b0, /*mode=*/0, g, FOURH, nullptr, 0, FOURH);
        lstm_pointwise<<<CDIV(B_ * H_, TB), TB, 0, stream>>>(
            g, hcd + H_, TWOH, hc0, hc0b);

        // -------- layer 1 --------
        gemm_wmma_bf16<<<gEnc, TB, 0, stream>>>(
            hc1b, TWOH, WeT, TWOH, nullptr, 0, nullptr, 0,
            be, /*mode=*/1, nullptr, 0, q, BN_, BN_);
        gemm_wmma_bf16<<<gDec, TB, 0, stream>>>(
            q, BN_, WdT, BN_, nullptr, 0, nullptr, 0,
            bd, /*mode=*/0, hcd, TWOH, hcdb, TWOH, TWOH);
        // gates: g = h0 @ Wih1 + h1' @ Whh1 + b1   (h0 = cols [0,H) of hc0b)
        gemm_wmma_bf16<<<gGates, TB, 0, stream>>>(
            hc0b, TWOH, Wih1T, H_, hcdb, TWOH, Whh1T, H_,
            b1, /*mode=*/0, g, FOURH, nullptr, 0, FOURH);
        lstm_pointwise<<<CDIV(B_ * H_, TB), TB, 0, stream>>>(
            g, hcd + H_, TWOH, hc1, hc1b);

        // -------- output head --------
        if (t == SEQ_ - 1)
            dense_head<<<CDIV(B_ * OUT_, TB), TB, 0, stream>>>(
                hc1, Wf, bfv, outp);                                   // slot 0
        else if (la)
            dense_head<<<CDIV(B_ * OUT_, TB), TB, 0, stream>>>(
                hc1, Wf, bfv, outp + (size_t)(t - SEQ_ + 1) * OUT_);   // slot t-127
    }
    #undef CDIV
}
